// GCNLayer_20289425507106
// MI455X (gfx1250) — matrix operand order repaired and checked by reading, not compiled, not run on hardware
//
#include <hip/hip_runtime.h>

#define Dm   128
#define Nn   384
#define Bb   2
#define Kk   16
#define EPSf 1e-5f

typedef __bf16 bf16_t;
typedef __bf16 v16bf __attribute__((ext_vector_type(16)));
typedef __bf16 v8bf  __attribute__((ext_vector_type(8)));
typedef __bf16 v4bf  __attribute__((ext_vector_type(4)));
typedef float  v8f   __attribute__((ext_vector_type(8)));

// ---------------- WMMA helpers (CDNA5 wave32, 16x16x32 bf16 -> f32) ----------------

static __device__ __forceinline__ v8f wmma_bf16(v16bf a, v16bf b, v8f c) {
    // 8 args: (neg_a, A, neg_b, B, c_mod, C, reuse_a, reuse_b)
    return __builtin_amdgcn_wmma_f32_16x16x32_bf16(false, a, false, b, (short)0, c, false, false);
}

// A fragment: 16x32 bf16 tile slice from row-major [16][128] LDS tile.
// lanes 0-15 hold M=lane, K = ks+0..15; lanes 16-31 hold K = ks+16..31.
static __device__ __forceinline__ v16bf load_afrag(const bf16_t* t, int lane, int ks) {
    const int m  = lane & 15;
    const int k0 = ks + ((lane >> 4) << 4);
    union { v16bf v; v8bf h[2]; } u;
    u.h[0] = *(const v8bf*)(t + m * Dm + k0);
    u.h[1] = *(const v8bf*)(t + m * Dm + k0 + 8);
    return u.v;
}

// B fragment: 32x16 slice of W^T, i.e. B[k][n] = W[n][k], W row-major [out=128][in=128].
// lanes 0-15 hold K = ks+0..15, lanes 16-31 hold K = ks+16..31; N = lane%16.
static __device__ __forceinline__ v16bf load_bfrag(const bf16_t* W, int lane, int nt, int ks) {
    const int n  = nt * 16 + (lane & 15);
    const int k0 = ks + ((lane >> 4) << 4);
    return *(const v16bf*)(W + n * Dm + k0);
}

// ---------------- prep kernels ----------------

// Convert the 5 edge-path weight matrices to bf16 [out][in] into ws, and pack the
// 7 f32 param vectors (We_b, Vin_b, Ve_b, ln1g, ln1b, ln2g, ln2b).
__global__ __launch_bounds__(256) void prep_w(
    const float* __restrict__ W1, const float* __restrict__ We,
    const float* __restrict__ Vin, const float* __restrict__ Ve,
    const float* __restrict__ Web, const float* __restrict__ Vinb,
    const float* __restrict__ Veb,
    const float* __restrict__ l1g, const float* __restrict__ l1b,
    const float* __restrict__ l2g, const float* __restrict__ l2b,
    bf16_t* __restrict__ wbf, float* __restrict__ pbuf)
{
    const int i = blockIdx.x * 256 + threadIdx.x;
    if (i < 16384)              wbf[i] = (bf16_t)W1[i];
    else if (i < 32768)         wbf[i] = (bf16_t)We[i - 16384];
    else if (i < 49152)         wbf[i] = (bf16_t)Vin[i - 32768];
    else if (i < 65536) { const int j = i - 49152; wbf[i] = (bf16_t)Ve[(j >> 7) * 256 + (j & 127)]; }
    else if (i < 81920) { const int j = i - 65536; wbf[i] = (bf16_t)Ve[(j >> 7) * 256 + 128 + (j & 127)]; }
    else if (i < 81920 + 7 * Dm) {
        const int j = i - 81920;
        const int which = j >> 7, c = j & 127;
        const float* s;
        switch (which) {
            case 0: s = Web;  break;
            case 1: s = Vinb; break;
            case 2: s = Veb;  break;
            case 3: s = l1g;  break;
            case 4: s = l1b;  break;
            case 5: s = l2g;  break;
            default: s = l2b; break;
        }
        pbuf[j] = s[c];
    }
}

// w2x[b,n,:] = x[b,n]@W2^T + W2_b ; w3x[b,n,:] = x[b,n]@W3^T + W3_b  (f32)
__global__ __launch_bounds__(256) void prep_xw(
    const float* __restrict__ x,
    const float* __restrict__ W2, const float* __restrict__ W2b,
    const float* __restrict__ W3, const float* __restrict__ W3b,
    float* __restrict__ w2x, float* __restrict__ w3x)
{
    const int row = blockIdx.x;               // 0 .. B*N-1
    const int d   = threadIdx.x & (Dm - 1);
    const int h   = threadIdx.x >> 7;         // 0 -> W2 path, 1 -> W3 path
    const float* xr = x + (size_t)row * Dm;
    const float* w  = (h ? W3 : W2) + d * Dm;
    float s = (h ? W3b : W2b)[d];
#pragma unroll 8
    for (int j = 0; j < Dm; ++j) s += xr[j] * w[j];
    (h ? w3x : w2x)[(size_t)row * Dm + d] = s;
}

// ---------------- node path (~2% of FLOPs): one block of 128 threads per node ----------------

static __device__ __forceinline__ float blockSum128(float v, float* red) {
#pragma unroll
    for (int m = 16; m >= 1; m >>= 1) v += __shfl_xor(v, m, 32);
    const int t = threadIdx.x;
    if ((t & 31) == 0) red[t >> 5] = v;
    __syncthreads();
    const float s = red[0] + red[1] + red[2] + red[3];
    __syncthreads();
    return s;
}

__global__ __launch_bounds__(128) void node_kernel(
    const float* __restrict__ x, const int* __restrict__ nbi,
    const float* __restrict__ wq, const float* __restrict__ bq,
    const float* __restrict__ wk, const float* __restrict__ bk,
    const float* __restrict__ wv, const float* __restrict__ bv,
    const float* __restrict__ wo, const float* __restrict__ bo,
    const float* __restrict__ Wn, const float* __restrict__ Wnb,
    const float* __restrict__ Vin, const float* __restrict__ Vinb,
    const float* __restrict__ Vn, const float* __restrict__ Vnb,
    const float* __restrict__ g1, const float* __restrict__ b1,
    const float* __restrict__ g2, const float* __restrict__ b2,
    float* __restrict__ out)
{
    __shared__ float xi[Dm], qb[Dm], att[Dm], hnb[Dm], vin[Dm];
    __shared__ float nbx[Kk][Dm], kmat[Kk][Dm], vmat[Kk][Dm];
    __shared__ float sc[Kk], red[4];

    const int row = blockIdx.x;          // b*N + n
    const int d   = threadIdx.x;
    const int b   = row / Nn;

    xi[d] = x[(size_t)row * Dm + d];
#pragma unroll
    for (int k2 = 0; k2 < Kk; ++k2) {
        const int nb = nbi[row * Kk + k2];
        nbx[k2][d] = x[((size_t)b * Nn + nb) * Dm + d];
    }
    __syncthreads();

    { // q = x @ wq^T + bq
        float s = bq[d]; const float* w = wq + d * Dm;
        for (int j = 0; j < Dm; ++j) s += xi[j] * w[j];
        qb[d] = s;
    }
    for (int k2 = 0; k2 < Kk; ++k2) {  // k,v per neighbor
        float sk = bk[d], sv = bv[d];
        const float* wkd = wk + d * Dm;
        const float* wvd = wv + d * Dm;
        for (int j = 0; j < Dm; ++j) { const float xv = nbx[k2][j]; sk += xv * wkd[j]; sv += xv * wvd[j]; }
        kmat[k2][d] = sk; vmat[k2][d] = sv;
    }
    __syncthreads();
    if (d < Kk) {
        float s = 0.f;
        for (int j = 0; j < Dm; ++j) s += qb[j] * kmat[d][j];
        sc[d] = s * 0.088388347648318447f;   // 1/sqrt(128)
    }
    __syncthreads();
    // softmax over K (computed redundantly by every thread)
    float mx = sc[0];
#pragma unroll
    for (int k2 = 1; k2 < Kk; ++k2) mx = fmaxf(mx, sc[k2]);
    float aw[Kk]; float den = 0.f;
#pragma unroll
    for (int k2 = 0; k2 < Kk; ++k2) { aw[k2] = __expf(sc[k2] - mx); den += aw[k2]; }
    const float inv = 1.f / den;
    float av = 0.f;
#pragma unroll
    for (int k2 = 0; k2 < Kk; ++k2) av += aw[k2] * vmat[k2][d];
    att[d] = av * inv;
    __syncthreads();
    { // out-proj
        float s = bo[d]; const float* w = wo + d * Dm;
        for (int j = 0; j < Dm; ++j) s += att[j] * w[j];
        qb[d] = s;
    }
    __syncthreads();
    float t1;
    { // node residual 1 + LN1
        float s = Wnb[d]; const float* w = Wn + d * Dm;
        for (int j = 0; j < Dm; ++j) s += qb[j] * w[j];
        t1 = xi[d] + fmaxf(s, 0.f);
    }
    float mu  = blockSum128(t1, red) * (1.f / Dm);
    float var = blockSum128(t1 * t1, red) * (1.f / Dm) - mu * mu;
    const float hn = (t1 - mu) * rsqrtf(var + EPSf) * g1[d] + b1[d];
    hnb[d] = hn;
    {
        float s = Vinb[d]; const float* w = Vin + d * Dm;
        for (int j = 0; j < Dm; ++j) s += xi[j] * w[j];
        vin[d] = s;
    }
    __syncthreads();
    float t2;
    { // cat -> V_node (Vn is [128][256])
        float s = Vnb[d]; const float* w = Vn + d * (2 * Dm);
        for (int j = 0; j < Dm; ++j) s += vin[j] * w[j];
        for (int j = 0; j < Dm; ++j) s += hnb[j] * w[Dm + j];
        t2 = hn + fmaxf(s, 0.f);
    }
    float mu2  = blockSum128(t2, red) * (1.f / Dm);
    float var2 = blockSum128(t2 * t2, red) * (1.f / Dm) - mu2 * mu2;
    out[(size_t)row * Dm + d] = (t2 - mu2) * rsqrtf(var2 + EPSf) * g2[d] + b2[d];
}

// ---------------- fused edge kernel (WMMA) ----------------
// dynamic LDS layout (bytes):
#define SM_WBF 0                               // 5 * 128*128 bf16 = 163840
#define SM_PAR 163840                          // 7 * 128 f32     = 3584
#define SM_ES  167424                          // 8 waves * 16*128 bf16 = 32768
#define SM_TS  200192                          // 8 waves * 16*128 bf16 = 32768
#define SM_RA  232960                          // 8 waves * 16*128 f32  = 65536
#define SM_EDGE_BYTES 298496
// weight element offsets within Wl:
#define WOFF_W1  0
#define WOFF_WE  16384
#define WOFF_VIN 32768
#define WOFF_VA  49152
#define WOFF_VB  65536

__global__ __launch_bounds__(256, 1) void edge_kernel(
    const float* __restrict__ eg,
    const float* __restrict__ w2x,
    const float* __restrict__ w3x,
    const float* __restrict__ W1b,
    const bf16_t* __restrict__ wbf_g,
    const float* __restrict__ pbuf_g,
    float* __restrict__ oute)
{
    extern __shared__ char smem[];
    bf16_t* Wl = (bf16_t*)(smem + SM_WBF);
    float*  pr = (float* )(smem + SM_PAR);
    const int tid = threadIdx.x;

    // ---- cooperative load of all 5 weight matrices (bf16) + params into LDS ----
    {
        const uint4* s4 = (const uint4*)wbf_g;
        uint4* d4 = (uint4*)Wl;
#pragma unroll 4
        for (int i = tid; i < (5 * Dm * Dm) / 8; i += 256) d4[i] = s4[i];
        for (int i = tid; i < 7 * Dm; i += 256) pr[i] = pbuf_g[i];
    }
    __syncthreads();

    const int lane = tid & 31;
    const int wave = tid >> 5;
    const int nlo  = lane & 15;
    const int hi8  = (lane >> 4) << 3;   // 0 or 8 (row offset of this half-wave)

    bf16_t* eS = (bf16_t*)(smem + SM_ES) + wave * (16 * Dm);
    bf16_t* tS = (bf16_t*)(smem + SM_TS) + wave * (16 * Dm);
    float*  rA = (float* )(smem + SM_RA) + wave * (16 * Dm);

    const int totalTiles = (Bb * Nn * Nn) / 16;    // 18432
    const int stride     = gridDim.x * 8;

    for (int tile = blockIdx.x * 8 + wave; tile < totalTiles; tile += stride) {
        const int g0  = tile * 16;       // first flat edge row: g = (b*N+i)*N + j
        const int bi  = g0 / Nn;         // b*N + i  (tiles never cross i: 16 | 384)
        const int j0  = g0 - bi * Nn;
        const int b   = bi / Nn;
        const int bj0 = b * Nn + j0;

        if (tile + stride < totalTiles)
            __builtin_prefetch(eg + (size_t)(tile + stride) * 16 * Dm, 0, 1);

        // ---- stage e tile (f32 -> bf16 -> LDS) and per-row add term (W1_b + W2x_i + W3x_j) ----
        const float4* ev = (const float4*)(eg + (size_t)g0 * Dm);
#pragma unroll 4
        for (int c = 0; c < 16; ++c) {
            const int idx = c * 32 + lane;           // 512 float4 = 16x128 f32
            const float4 f = ev[idx];
            v4bf pk = { (bf16_t)f.x, (bf16_t)f.y, (bf16_t)f.z, (bf16_t)f.w };
            *(v4bf*)(eS + (idx << 2)) = pk;
            const int m  = idx >> 5;
            const int n4 = (idx & 31) << 2;
            const float4 a2 = *(const float4*)(w2x + bi * Dm + n4);
            const float4 a3 = *(const float4*)(w3x + (size_t)(bj0 + m) * Dm + n4);
            const float4 bb = *(const float4*)(W1b + n4);
            float4 s;
            s.x = a2.x + a3.x + bb.x; s.y = a2.y + a3.y + bb.y;
            s.z = a2.z + a3.z + bb.z; s.w = a2.w + a3.w + bb.w;
            *(float4*)(rA + m * Dm + n4) = s;
        }
        asm volatile("s_wait_dscnt 0" ::: "memory");

        // ---- A fragments of e (reused by GEMM1 and GEMM3) ----
        v16bf ae[4];
#pragma unroll
        for (int ks = 0; ks < 4; ++ks) ae[ks] = load_afrag(eS, lane, ks * 32);

        // ---- GEMM1: pre = e @ W1^T + rowAdd  ->  tS (bf16) ----
#pragma unroll
        for (int t = 0; t < 8; ++t) {
            v8f acc = {0.f, 0.f, 0.f, 0.f, 0.f, 0.f, 0.f, 0.f};
#pragma unroll
            for (int ks = 0; ks < 4; ++ks)
                acc = wmma_bf16(ae[ks], load_bfrag(Wl + WOFF_W1, lane, t, ks * 32), acc);
            const int n = t * 16 + nlo;
#pragma unroll
            for (int r = 0; r < 8; ++r)
                tS[(r + hi8) * Dm + n] = (bf16_t)(acc[r] + rA[(r + hi8) * Dm + n]);
        }
        asm volatile("s_wait_dscnt 0" ::: "memory");

        // ---- GEMM2: hnb = LN1( e + relu(pre @ We^T + We_b) ) ----
        v16bf ap[4];
#pragma unroll
        for (int ks = 0; ks < 4; ++ks) ap[ks] = load_afrag(tS, lane, ks * 32);
        v8f hnb[8];
#pragma unroll
        for (int t = 0; t < 8; ++t) {
            v8f acc = {0.f, 0.f, 0.f, 0.f, 0.f, 0.f, 0.f, 0.f};
#pragma unroll
            for (int ks = 0; ks < 4; ++ks)
                acc = wmma_bf16(ap[ks], load_bfrag(Wl + WOFF_WE, lane, t, ks * 32), acc);
            const int n  = t * 16 + nlo;
            const float bn = pr[0 * Dm + n];
#pragma unroll
            for (int r = 0; r < 8; ++r) {
                const float h = fmaxf(acc[r] + bn, 0.f);
                hnb[t][r] = (float)eS[(r + hi8) * Dm + n] + h;
            }
        }
        // LayerNorm 1 over the 128 channels of each of the 16 rows
        float mu[8], rsd[8];
#pragma unroll
        for (int r = 0; r < 8; ++r) {
            float s = 0.f, s2 = 0.f;
#pragma unroll
            for (int t = 0; t < 8; ++t) { const float v = hnb[t][r]; s += v; s2 += v * v; }
#pragma unroll
            for (int msk = 8; msk >= 1; msk >>= 1) { s += __shfl_xor(s, msk, 32); s2 += __shfl_xor(s2, msk, 32); }
            const float m_ = s * (1.f / Dm);
            mu[r]  = m_;
            rsd[r] = rsqrtf(s2 * (1.f / Dm) - m_ * m_ + EPSf);
        }
#pragma unroll
        for (int t = 0; t < 8; ++t) {
            const int n = t * 16 + nlo;
            const float gg = pr[3 * Dm + n], bb = pr[4 * Dm + n];
#pragma unroll
            for (int r = 0; r < 8; ++r) {
                const float v = (hnb[t][r] - mu[r]) * rsd[r] * gg + bb;
                hnb[t][r] = v;
                tS[(r + hi8) * Dm + n] = (bf16_t)v;     // h_nb (bf16) for GEMM4
            }
        }

        // ---- GEMM3: vin = e @ Vin^T + Vin_b  ->  eS (bf16, e no longer needed in LDS) ----
#pragma unroll
        for (int t = 0; t < 8; ++t) {
            v8f acc = {0.f, 0.f, 0.f, 0.f, 0.f, 0.f, 0.f, 0.f};
#pragma unroll
            for (int ks = 0; ks < 4; ++ks)
                acc = wmma_bf16(ae[ks], load_bfrag(Wl + WOFF_VIN, lane, t, ks * 32), acc);
            const int n  = t * 16 + nlo;
            const float bn = pr[1 * Dm + n];
#pragma unroll
            for (int r = 0; r < 8; ++r)
                eS[(r + hi8) * Dm + n] = (bf16_t)(acc[r] + bn);
        }
        asm volatile("s_wait_dscnt 0" ::: "memory");

        // ---- GEMM4: h2 = relu([vin, hnb] @ Ve^T + Ve_b) ; out = LN2(hnb + h2) ----
        v16bf avf[4], ahf[4];
#pragma unroll
        for (int ks = 0; ks < 4; ++ks) {
            avf[ks] = load_afrag(eS, lane, ks * 32);
            ahf[ks] = load_afrag(tS, lane, ks * 32);
        }
#pragma unroll
        for (int t = 0; t < 8; ++t) {
            v8f acc = {0.f, 0.f, 0.f, 0.f, 0.f, 0.f, 0.f, 0.f};
#pragma unroll
            for (int ks = 0; ks < 4; ++ks)
                acc = wmma_bf16(avf[ks], load_bfrag(Wl + WOFF_VA, lane, t, ks * 32), acc);
#pragma unroll
            for (int ks = 0; ks < 4; ++ks)
                acc = wmma_bf16(ahf[ks], load_bfrag(Wl + WOFF_VB, lane, t, ks * 32), acc);
            const int n  = t * 16 + nlo;
            const float bn = pr[2 * Dm + n];
#pragma unroll
            for (int r = 0; r < 8; ++r)
                hnb[t][r] += fmaxf(acc[r] + bn, 0.f);   // residual + relu, in place
        }
        // LayerNorm 2 + store h_edge
#pragma unroll
        for (int r = 0; r < 8; ++r) {
            float s = 0.f, s2 = 0.f;
#pragma unroll
            for (int t = 0; t < 8; ++t) { const float v = hnb[t][r]; s += v; s2 += v * v; }
#pragma unroll
            for (int msk = 8; msk >= 1; msk >>= 1) { s += __shfl_xor(s, msk, 32); s2 += __shfl_xor(s2, msk, 32); }
            const float m_ = s * (1.f / Dm);
            mu[r]  = m_;
            rsd[r] = rsqrtf(s2 * (1.f / Dm) - m_ * m_ + EPSf);
        }
#pragma unroll
        for (int t = 0; t < 8; ++t) {
            const int n = t * 16 + nlo;
            const float gg = pr[5 * Dm + n], bb = pr[6 * Dm + n];
#pragma unroll
            for (int r = 0; r < 8; ++r)
                oute[(size_t)(g0 + r + hi8) * Dm + n] = (hnb[t][r] - mu[r]) * rsd[r] * gg + bb;
        }
    }
}

// ---------------- host launcher ----------------

extern "C" void kernel_launch(void* const* d_in, const int* in_sizes, int n_in,
                              void* d_out, int out_size, void* d_ws, size_t ws_size,
                              hipStream_t stream)
{
    (void)in_sizes; (void)n_in; (void)out_size; (void)ws_size;

    const float* x      = (const float*)d_in[0];
    const float* e      = (const float*)d_in[1];
    const int*   nbi    = (const int*  )d_in[2];
    const float* wq     = (const float*)d_in[3];
    const float* wk     = (const float*)d_in[4];
    const float* wv     = (const float*)d_in[5];
    const float* wo     = (const float*)d_in[6];
    const float* bq     = (const float*)d_in[7];
    const float* bk     = (const float*)d_in[8];
    const float* bv     = (const float*)d_in[9];
    const float* bo     = (const float*)d_in[10];
    const float* Wn_w   = (const float*)d_in[11];
    const float* Wn_b   = (const float*)d_in[12];
    const float* Vnin_w = (const float*)d_in[13];
    const float* Vnin_b = (const float*)d_in[14];
    const float* Vn_w   = (const float*)d_in[15];
    const float* Vn_b   = (const float*)d_in[16];
    const float* ln1n_g = (const float*)d_in[17];
    const float* ln1n_b = (const float*)d_in[18];
    const float* ln2n_g = (const float*)d_in[19];
    const float* ln2n_b = (const float*)d_in[20];
    const float* We_w   = (const float*)d_in[21];
    const float* We_b   = (const float*)d_in[22];
    const float* Vein_w = (const float*)d_in[23];
    const float* Vein_b = (const float*)d_in[24];
    const float* Ve_w   = (const float*)d_in[25];
    const float* Ve_b   = (const float*)d_in[26];
    const float* W1_w   = (const float*)d_in[27];
    const float* W1_b   = (const float*)d_in[28];
    const float* W2_w   = (const float*)d_in[29];
    const float* W2_b   = (const float*)d_in[30];
    const float* W3_w   = (const float*)d_in[31];
    const float* W3_b   = (const float*)d_in[32];
    const float* l1g    = (const float*)d_in[33];
    const float* l1b    = (const float*)d_in[34];
    const float* l2g    = (const float*)d_in[35];
    const float* l2b    = (const float*)d_in[36];

    // workspace layout
    char* ws = (char*)d_ws;
    bf16_t* wbf = (bf16_t*)(ws + 0);                 // 5*128*128 bf16 = 163840 B
    float*  pbf = (float* )(ws + 163840);            // 7*128 f32      = 3584 B
    float*  w2x = (float* )(ws + 167424);            // B*N*128 f32    = 393216 B
    float*  w3x = (float* )(ws + 167424 + 393216);   // B*N*128 f32    = 393216 B

    prep_w<<<324, 256, 0, stream>>>(W1_w, We_w, Vein_w, Ve_w, We_b, Vein_b, Ve_b,
                                    l1g, l1b, l2g, l2b, wbf, pbf);
    prep_xw<<<Bb * Nn, 256, 0, stream>>>(x, W2_w, W2_b, W3_w, W3_b, w2x, w3x);

    node_kernel<<<Bb * Nn, 128, 0, stream>>>(x, nbi, wq, bq, wk, bk, wv, bv, wo, bo,
                                             Wn_w, Wn_b, Vnin_w, Vnin_b, Vn_w, Vn_b,
                                             ln1n_g, ln1n_b, ln2n_g, ln2n_b,
                                             (float*)d_out);

    float* oute = (float*)d_out + (size_t)Bb * Nn * Dm;
    edge_kernel<<<576, 256, SM_EDGE_BYTES, stream>>>(e, w2x, w3x, W1_b, wbf, pbf, oute);
}
